// GenePathwayTransformerEncoder_83150566851031
// MI455X (gfx1250) — matrix operand order repaired and checked by reading, not compiled, not run on hardware
//
#include <hip/hip_runtime.h>
#include <hip/hip_bf16.h>

// Problem constants (match reference)
#define BB 4
#define GG 2048
#define PP 512
#define DD 512
#define HH 8
#define LL 4
#define DHH 64

typedef __attribute__((ext_vector_type(16))) __bf16 v16bf;
typedef __attribute__((ext_vector_type(8)))  __bf16 v8bf;
typedef __attribute__((ext_vector_type(8)))  float  v8f;
typedef __attribute__((ext_vector_type(4)))  unsigned int v4u;
typedef __attribute__((ext_vector_type(8)))  int v8i;
typedef __attribute__((ext_vector_type(4)))  int v4i;

__device__ __forceinline__ unsigned short f2bf(float f) {
  unsigned int u = __float_as_uint(f);
  u += 0x7FFFu + ((u >> 16) & 1u);           // round-to-nearest-even
  return (unsigned short)(u >> 16);
}
__device__ __forceinline__ float bf2f(unsigned short s) {
  return __uint_as_float(((unsigned int)s) << 16);
}

// A-matrix fragment (16x32 bf16): lane m = lane&15.
// lanes 0-15: K {k0..k0+7, k0+16..k0+23}; lanes 16-31: +8 on both chunks.
__device__ __forceinline__ v16bf load_a_frag(const unsigned short* p /* &A[m*ld + k0] */, int hi) {
  union { v16bf v; v8bf h[2]; } u;
  u.h[0] = *(const v8bf*)(p + hi);
  u.h[1] = *(const v8bf*)(p + 16 + hi);
  return u.v;
}
// B-matrix fragment (32x16 bf16): lane n = lane&15; K {hi..hi+7, 16+hi..16+hi+7} strided by ldn.
__device__ __forceinline__ v16bf load_b_frag(const unsigned short* p /* &Bt[k0*ldn + n0 + (lane&15)] */, int ldn, int hi) {
  union { v16bf v; unsigned short s[16]; } u;
#pragma unroll
  for (int i = 0; i < 8; ++i) {
    u.s[i] = p[(size_t)(hi + i) * ldn];
    u.s[8 + i] = p[(size_t)(16 + hi + i) * ldn];
  }
  return u.v;
}

// ---------------- Tensor Data Mover: 2D tile Global -> LDS ----------------
// Packs D# per CDNA5 ISA ch.8: group0 {count=1, lds_addr, global_addr, type=2},
// group1 {data_size=2B, tensor_dim0/1, tile_dim0/1, dim0_stride}. Issued by one
// wave (EXEC-independent); completion tracked with TENSORcnt.
__device__ __forceinline__ void tdm_load_tile_2d(unsigned short* lds_dst, const void* gsrc,
                                                 unsigned int tile_d0, unsigned int tile_d1,
                                                 unsigned int tensor_d0, unsigned int tensor_d1,
                                                 unsigned int row_stride_elems, int lane) {
#if __has_builtin(__builtin_amdgcn_tensor_load_to_lds)
  (void)lane;
  unsigned long long la = (unsigned long long)(size_t)lds_dst;
  unsigned long long ga = (unsigned long long)(size_t)gsrc;
  v4u g0;
  g0[0] = 1u;                                   // count=1 (valid descriptor)
  g0[1] = (unsigned int)la;                     // lds_addr (bytes), bits 63:32
  g0[2] = (unsigned int)ga;                     // global_addr[31:0]
  g0[3] = (unsigned int)((ga >> 32) & 0x01FFFFFFu) | 0x80000000u; // addr[56:32] | type=2
  v8i g1;
  g1[0] = (int)(1u << 16);                      // data_size=1 -> 2 bytes
  g1[1] = (int)((tensor_d0 & 0xFFFFu) << 16);                               // dim0[15:0]
  g1[2] = (int)(((tensor_d0 >> 16) & 0xFFFFu) | ((tensor_d1 & 0xFFFFu) << 16));
  g1[3] = (int)(((tensor_d1 >> 16) & 0xFFFFu) | ((tile_d0 & 0xFFFFu) << 16)); // tile_dim0
  g1[4] = (int)(tile_d1 & 0xFFFFu);             // tile_dim1 (tile_dim2 = 0)
  g1[5] = (int)row_stride_elems;                // tensor_dim0_stride[31:0]
  g1[6] = 0;
  g1[7] = 0;
  v4i gz4 = {0, 0, 0, 0};
  v8i gz8 = {0, 0, 0, 0, 0, 0, 0, 0};
  __builtin_amdgcn_tensor_load_to_lds(g0, g1, gz4, gz4, gz8, 0);
#else
  // Wave-cooperative fallback copy
  const unsigned short* src = (const unsigned short*)gsrc;
  const unsigned int tot = tile_d0 * tile_d1;
  for (unsigned int i = lane; i < tot; i += 32) {
    unsigned int r = i / tile_d0, c = i - r * tile_d0;
    lds_dst[i] = src[(size_t)r * row_stride_elems + c];
  }
  (void)tensor_d0; (void)tensor_d1;
#endif
}
__device__ __forceinline__ void tdm_wait() {
#if __has_builtin(__builtin_amdgcn_tensor_load_to_lds)
  __builtin_amdgcn_s_wait_tensorcnt(0);
#endif
}

// ---------------- embedding / conversion kernels ----------------

__global__ void __launch_bounds__(256)
embed_genes_kernel(const int* __restrict__ gi, const float* __restrict__ ev,
                   const float* __restrict__ gemb, unsigned short* __restrict__ X) {
  size_t i = (size_t)blockIdx.x * 256 + threadIdx.x;          // B*G*D
  int d = (int)(i % DD);
  size_t t = i / DD;
  int g = (int)(t % GG);
  int b = (int)(t / GG);
  int row = gi[b * GG + g];
  float v = gemb[(size_t)row * DD + d] * ev[b * GG + g];
  X[i] = f2bf(v);
}

__global__ void __launch_bounds__(256)
embed_path_kernel(const int* __restrict__ pi, const float* __restrict__ pemb,
                  unsigned short* __restrict__ PE) {
  size_t i = (size_t)blockIdx.x * 256 + threadIdx.x;          // B*P*D
  int d = (int)(i % DD);
  size_t t = i / DD;
  int p = (int)(t % PP);
  int row = pi[(t / PP) * PP + p];
  PE[i] = f2bf(pemb[(size_t)row * DD + d]);
}

// Wt[l][mat][k][n] = W[l][mat][n][k] in bf16 (K-major operand for WMMA B frags)
__global__ void __launch_bounds__(256)
wt_kernel(const float* __restrict__ Wq, const float* __restrict__ Wk,
          const float* __restrict__ Wv, const float* __restrict__ Wo,
          unsigned short* __restrict__ Wt) {
  size_t i = (size_t)blockIdx.x * 256 + threadIdx.x;          // L*4*D*D
  int n = (int)(i % DD);
  size_t t = i / DD;
  int k = (int)(t % DD);
  t /= DD;
  int mat = (int)(t % 4);
  int l = (int)(t / 4);
  const float* W = (mat == 0 ? Wq : (mat == 1 ? Wk : (mat == 2 ? Wv : Wo))) + (size_t)l * DD * DD;
  Wt[i] = f2bf(W[(size_t)n * DD + k]);
}

// ---------------- WMMA GEMM (K=N=512): C = A @ W^T + bias ----------------
// Block = 32 rows x 256 cols; wave computes a 32x32 tile (2 A frags x 2 B frags,
// each B frag reused twice -> half the global traffic per WMMA).
// A: [M,512] bf16 row-major (TDM-staged 32x512 tile into LDS).
// Bt: [512,512] bf16 K-major. mode 0: C row-major [M,512] bf16.
// mode 1: scatter into KT layout [B,H,DH,P].
__global__ void __launch_bounds__(256)
gemm_bf16_kernel(const unsigned short* __restrict__ A,
                 const unsigned short* __restrict__ Bt,
                 const float* __restrict__ bias,
                 unsigned short* __restrict__ C,
                 int mode) {
  constexpr int K = 512, N = 512;
  __shared__ unsigned short sAT[32][K];                       // 32 KB A tile

  const int tid  = threadIdx.x;
  const int wave = tid >> 5;
  const int lane = tid & 31;
  const int m0 = (blockIdx.x >> 1) * 32;
  const int ng = blockIdx.x & 1;                              // column half

  if (wave == 0) {
    tdm_load_tile_2d(&sAT[0][0], A + (size_t)m0 * K, K, 32, K, 32, K, lane);
    tdm_wait();
  }
  __syncthreads();

  const int n0 = ng * 256 + wave * 32;                        // 8 waves x 32 cols
  const int mA = lane & 15;
  const int hiA = (lane < 16) ? 0 : 8;
  const unsigned short* aRow0 = &sAT[mA][0];
  const unsigned short* aRow1 = &sAT[16 + mA][0];
  const unsigned short* bptr = Bt + (lane & 15) + n0;         // column n=lane&15
  const int bstep = 32 * N;                                   // advance 32 K-rows

  v8f acc[4] = {};                                            // [rowgrp*2 + coltile]
  v16bf bcur[2], bnxt[2];
#pragma unroll
  for (int j = 0; j < 2; ++j) bcur[j] = load_b_frag(bptr + j * 16, N, hiA);
  bptr += bstep;

#pragma unroll
  for (int k0 = 0; k0 < K - 32; k0 += 32) {
#pragma unroll
    for (int j = 0; j < 2; ++j) bnxt[j] = load_b_frag(bptr + j * 16, N, hiA);
    bptr += bstep;
    v16bf a0 = load_a_frag(aRow0 + k0, hiA);
    v16bf a1 = load_a_frag(aRow1 + k0, hiA);
    acc[0] = __builtin_amdgcn_wmma_f32_16x16x32_bf16(false, a0, false, bcur[0], (short)0, acc[0], false, false);
    acc[1] = __builtin_amdgcn_wmma_f32_16x16x32_bf16(false, a0, false, bcur[1], (short)0, acc[1], false, false);
    acc[2] = __builtin_amdgcn_wmma_f32_16x16x32_bf16(false, a1, false, bcur[0], (short)0, acc[2], false, false);
    acc[3] = __builtin_amdgcn_wmma_f32_16x16x32_bf16(false, a1, false, bcur[1], (short)0, acc[3], false, false);
#pragma unroll
    for (int j = 0; j < 2; ++j) bcur[j] = bnxt[j];
  }
  {
    v16bf a0 = load_a_frag(aRow0 + (K - 32), hiA);
    v16bf a1 = load_a_frag(aRow1 + (K - 32), hiA);
    acc[0] = __builtin_amdgcn_wmma_f32_16x16x32_bf16(false, a0, false, bcur[0], (short)0, acc[0], false, false);
    acc[1] = __builtin_amdgcn_wmma_f32_16x16x32_bf16(false, a0, false, bcur[1], (short)0, acc[1], false, false);
    acc[2] = __builtin_amdgcn_wmma_f32_16x16x32_bf16(false, a1, false, bcur[0], (short)0, acc[2], false, false);
    acc[3] = __builtin_amdgcn_wmma_f32_16x16x32_bf16(false, a1, false, bcur[1], (short)0, acc[3], false, false);
  }

  const int mC = (lane < 16) ? 0 : 8;
  const int nC = lane & 15;
#pragma unroll
  for (int ct = 0; ct < 2; ++ct) {
    const int n = n0 + ct * 16 + nC;
    const float bi = bias[n];
#pragma unroll
    for (int rg = 0; rg < 2; ++rg) {
#pragma unroll
      for (int r = 0; r < 8; ++r) {
        const int m = m0 + rg * 16 + mC + r;
        const float val = acc[rg * 2 + ct][r] + bi;
        if (mode == 0) {
          C[(size_t)m * N + n] = f2bf(val);
        } else {
          const int b = m / PP, p = m - b * PP;
          const int h = n / DHH, d = n - h * DHH;
          C[(((size_t)(b * HH + h)) * DHH + d) * PP + p] = f2bf(val);
        }
      }
    }
  }
}

// ---------------- fused attention for one (b, h, 16-gene tile) ----------------
// Q: [B*G, D] bf16.  KT: [B,H,DH,P] bf16.  V: [B*P, D] bf16.  Wmask: [G,P] f32.
// O: [B*G, D] bf16.
__global__ void __launch_bounds__(256)
attn_kernel(const unsigned short* __restrict__ Q,
            const unsigned short* __restrict__ KT,
            const unsigned short* __restrict__ V,
            const float* __restrict__ Wmask,
            unsigned short* __restrict__ O) {
  __shared__ float sS[16][PP];            // 32 KB scores
  __shared__ unsigned short sA[16][PP];   // 16 KB attn (bf16)
  __shared__ unsigned short sQ[16][DHH];  //  2 KB q tile (TDM-staged)
  __shared__ float sO[2][16][DHH];        //  8 KB split-K partials
  __shared__ float sRed[16][17];

  const int tid  = threadIdx.x;
  const int wave = tid >> 5;
  const int lane = tid & 31;
  const int ntile = GG / 16;
  const int gt = blockIdx.x % ntile;
  const int h  = (blockIdx.x / ntile) % HH;
  const int b  = blockIdx.x / (ntile * HH);
  const int g0 = gt * 16;

  const int mA = lane & 15;
  const int hiA = (lane < 16) ? 0 : 8;
  const int mC = (lane < 16) ? 0 : 8;
  const int nC = lane & 15;

  if (wave == 0) {
    tdm_load_tile_2d(&sQ[0][0], Q + ((size_t)(b * GG + g0)) * DD + h * DHH,
                     DHH, 16, DHH, 16, DD, lane);
    tdm_wait();
  }
  __syncthreads();

  // Phase 1: scores[16, 512] = (Q_tile @ K^T) * scale * sigmoid(W_mask)
  {
    const unsigned short* ktb = KT + ((size_t)(b * HH + h)) * DHH * PP;
    const int p0 = wave * 64;                               // 8 waves cover P=512
    const unsigned short* bp = ktb + (lane & 15) + p0;
    v16bf kb0[4], kb1[4];
#pragma unroll
    for (int j = 0; j < 4; ++j) kb0[j] = load_b_frag(bp + j * 16, PP, hiA);
#pragma unroll
    for (int j = 0; j < 4; ++j) kb1[j] = load_b_frag(bp + 32 * PP + j * 16, PP, hiA);
    v16bf a0 = load_a_frag(&sQ[mA][0], hiA);
    v16bf a1 = load_a_frag(&sQ[mA][32], hiA);
    v8f acc[4] = {};
#pragma unroll
    for (int j = 0; j < 4; ++j)
      acc[j] = __builtin_amdgcn_wmma_f32_16x16x32_bf16(false, a0, false, kb0[j],
                                                       (short)0, acc[j], false, false);
#pragma unroll
    for (int j = 0; j < 4; ++j)
      acc[j] = __builtin_amdgcn_wmma_f32_16x16x32_bf16(false, a1, false, kb1[j],
                                                       (short)0, acc[j], false, false);
#pragma unroll
    for (int j = 0; j < 4; ++j) {
      const int p = p0 + j * 16 + nC;
#pragma unroll
      for (int r = 0; r < 8; ++r) {
        const int m = mC + r;
        const float w = Wmask[(size_t)(g0 + m) * PP + p];
        const float gate = 1.0f / (1.0f + __expf(-w));
        sS[m][p] = acc[j][r] * 0.125f * gate;               // scale = 1/sqrt(64)
      }
    }
  }
  __syncthreads();

  // Phase 2: softmax over P per row (16 threads per row)
  const int row = tid >> 4;
  const int sub = tid & 15;
  float mx = -3.0e38f;
  for (int p = sub; p < PP; p += 16) mx = fmaxf(mx, sS[row][p]);
  sRed[row][sub] = mx;
  __syncthreads();
  if (sub == 0) {
    float m2 = -3.0e38f;
    for (int i = 0; i < 16; ++i) m2 = fmaxf(m2, sRed[row][i]);
    sRed[row][16] = m2;
  }
  __syncthreads();
  const float rmax = sRed[row][16];
  float sum = 0.0f;
  for (int p = sub; p < PP; p += 16) {
    float e = __expf(sS[row][p] - rmax);
    sS[row][p] = e;
    sum += e;
  }
  __syncthreads();
  sRed[row][sub] = sum;
  __syncthreads();
  if (sub == 0) {
    float s2 = 0.0f;
    for (int i = 0; i < 16; ++i) s2 += sRed[row][i];
    sRed[row][16] = 1.0f / s2;
  }
  __syncthreads();
  const float rinv = sRed[row][16];
  for (int p = sub; p < PP; p += 16) sA[row][p] = f2bf(sS[row][p] * rinv);
  __syncthreads();

  // Phase 3: O_tile[16,64] = attn[16,512] @ V[:,h*64:h*64+64]
  // All 8 waves: wave = {kh in 0..1} x {4 n-tiles}; split-K partials in LDS.
  {
    const int kh = wave >> 2;                 // K half: kh*256 .. kh*256+255
    const int n0 = (wave & 3) * 16;
    const unsigned short* aRow = &sA[mA][kh * 256];
    const unsigned short* vp = V + ((size_t)(b * PP + kh * 256)) * DD + h * DHH + n0 + (lane & 15);
    const int vstep = 32 * DD;
    v8f acc = {};
    v16bf bcur = load_b_frag(vp, DD, hiA);
    vp += vstep;
#pragma unroll
    for (int k0 = 0; k0 < 256 - 32; k0 += 32) {
      v16bf bnxt = load_b_frag(vp, DD, hiA);
      vp += vstep;
      v16bf a = load_a_frag(aRow + k0, hiA);
      acc = __builtin_amdgcn_wmma_f32_16x16x32_bf16(false, a, false, bcur,
                                                    (short)0, acc, false, false);
      bcur = bnxt;
    }
    {
      v16bf a = load_a_frag(aRow + (256 - 32), hiA);
      acc = __builtin_amdgcn_wmma_f32_16x16x32_bf16(false, a, false, bcur,
                                                    (short)0, acc, false, false);
    }
#pragma unroll
    for (int r = 0; r < 8; ++r) sO[kh][mC + r][n0 + nC] = acc[r];
  }
  __syncthreads();
  for (int idx = tid; idx < 16 * DHH; idx += 256) {
    const int m = idx >> 6, n = idx & 63;
    O[((size_t)(b * GG + g0 + m)) * DD + h * DHH + n] = f2bf(sO[0][m][n] + sO[1][m][n]);
  }
}

// ---------------- final reduction + output projection ----------------

__global__ void __launch_bounds__(256)
mean_kernel(const unsigned short* __restrict__ X, float* __restrict__ xm) {
  int i = blockIdx.x * 256 + threadIdx.x;         // B*D
  int b = i / DD, d = i % DD;
  float s = 0.0f;
  for (int g = 0; g < GG; ++g) s += bf2f(X[((size_t)(b * GG + g)) * DD + d]);
  xm[i] = s * (1.0f / (float)GG);
}

__global__ void __launch_bounds__(256)
out_kernel(const float* __restrict__ xm, const float* __restrict__ Wout,
           const float* __restrict__ bout, float* __restrict__ out) {
  int i = blockIdx.x * 256 + threadIdx.x;         // B*P
  int b = i / PP, p = i % PP;
  float s = bout[p];
  for (int d = 0; d < DD; ++d) s += xm[b * DD + d] * Wout[(size_t)p * DD + d];
  out[i] = s;
}

// ---------------- host-side orchestration ----------------

extern "C" void kernel_launch(void* const* d_in, const int* in_sizes, int n_in,
                              void* d_out, int out_size, void* d_ws, size_t ws_size,
                              hipStream_t stream) {
  const int*   gi   = (const int*)  d_in[0];
  const float* ev   = (const float*)d_in[1];
  const int*   pidx = (const int*)  d_in[2];
  const float* gemb = (const float*)d_in[3];
  const float* pemb = (const float*)d_in[4];
  const float* Wq = (const float*)d_in[5];
  const float* bq = (const float*)d_in[6];
  const float* Wk = (const float*)d_in[7];
  const float* bk = (const float*)d_in[8];
  const float* Wv = (const float*)d_in[9];
  const float* bv = (const float*)d_in[10];
  const float* Wo = (const float*)d_in[11];
  const float* bo = (const float*)d_in[12];
  const float* Wm = (const float*)d_in[13];
  const float* Wout = (const float*)d_in[14];
  const float* bout = (const float*)d_in[15];
  float* out = (float*)d_out;

  char* ws = (char*)d_ws;
  size_t off = 0;
  auto alloc = [&](size_t bytes) -> char* {
    char* p = ws + off;
    off += (bytes + 255) & ~(size_t)255;
    return p;
  };
  unsigned short* xb = (unsigned short*)alloc((size_t)BB * GG * DD * 2);         // x (bf16)
  unsigned short* pe = (unsigned short*)alloc((size_t)BB * PP * DD * 2);         // pathway emb
  unsigned short* qb = (unsigned short*)alloc((size_t)BB * GG * DD * 2);         // q
  unsigned short* kT = (unsigned short*)alloc((size_t)BB * HH * DHH * PP * 2);   // k transposed
  unsigned short* vb = (unsigned short*)alloc((size_t)BB * PP * DD * 2);         // v
  unsigned short* ob = (unsigned short*)alloc((size_t)BB * GG * DD * 2);         // attn out
  unsigned short* Wt = (unsigned short*)alloc((size_t)LL * 4 * DD * DD * 2);     // W^T (bf16)
  float* xm = (float*)alloc((size_t)BB * DD * 4);
  (void)ws_size; (void)in_sizes; (void)n_in; (void)out_size;

  dim3 blk(256);
  embed_genes_kernel<<<(BB * GG * DD) / 256, blk, 0, stream>>>(gi, ev, gemb, xb);
  embed_path_kernel<<<(BB * PP * DD) / 256, blk, 0, stream>>>(pidx, pemb, pe);
  wt_kernel<<<(LL * 4 * DD * DD) / 256, blk, 0, stream>>>(Wq, Wk, Wv, Wo, Wt);

  const size_t WSZ = (size_t)DD * DD;
  for (int l = 0; l < LL; ++l) {
    const unsigned short* Wqt = Wt + ((size_t)l * 4 + 0) * WSZ;
    const unsigned short* Wkt = Wt + ((size_t)l * 4 + 1) * WSZ;
    const unsigned short* Wvt = Wt + ((size_t)l * 4 + 2) * WSZ;
    const unsigned short* Wot = Wt + ((size_t)l * 4 + 3) * WSZ;
    gemm_bf16_kernel<<<(BB * GG / 32) * 2, blk, 0, stream>>>(xb, Wqt, bq + l * DD, qb, 0);
    gemm_bf16_kernel<<<(BB * PP / 32) * 2, blk, 0, stream>>>(pe, Wkt, bk + l * DD, kT, 1);
    gemm_bf16_kernel<<<(BB * PP / 32) * 2, blk, 0, stream>>>(pe, Wvt, bv + l * DD, vb, 0);
    attn_kernel<<<BB * HH * (GG / 16), blk, 0, stream>>>(qb, kT, vb, Wm, ob);
    gemm_bf16_kernel<<<(BB * GG / 32) * 2, blk, 0, stream>>>(ob, Wot, bo + l * DD, xb, 0);
  }
  mean_kernel<<<(BB * DD) / 256, blk, 0, stream>>>(xb, xm);
  out_kernel<<<(BB * PP) / 256, blk, 0, stream>>>(xm, Wout, bout, out);
}
